// VectorQuantizer_38242388804087
// MI455X (gfx1250) — matrix operand order repaired and checked by reading, not compiled, not run on hardware
//
#include <hip/hip_runtime.h>
#include <hip/hip_fp16.h>

typedef __attribute__((ext_vector_type(16))) _Float16 v16h;
typedef __attribute__((ext_vector_type(8)))  float    v8f;

#define VQ_N_E   1024
#define VQ_EDIM  64
#define VQ_NTOK  131072
#define VQ_BETA  0.4f

// Workspace layout (bytes):
//   [0,      131072)  cb16      : _Float16[1024*64]  (f16 codebook, row-major)
//   [131072, 135168)  c2        : float[1024]        (|c|^2 per code)
//   [135168, 139264)  hist      : unsigned[1024]     (codebook usage counts)
//   [139264, 141312)  blockErr  : float[512]         (per-block SSE partials)

// Dynamic LDS layout (per block, 137216 B):
//   [0,      131072)  cbL  : f16 codebook copy (async-staged)
//   [131072, 135168)  c2L  : float[1024]
//   [135168, 136192)  sIdx : int[256]   (chosen code per local token)
//   [136192, 137216)  sErr : float[256]
#define VQ_LDS_BYTES (131072 + 4096 + 1024 + 1024)

// ---------------------------------------------------------------------------
// Kernel 0: codebook prep — f32 -> f16 copy, |c|^2, zero histogram
// ---------------------------------------------------------------------------
__global__ void vq_prep(const float* __restrict__ cb,
                        _Float16* __restrict__ cb16,
                        float* __restrict__ c2,
                        unsigned* __restrict__ hist) {
    int j = blockIdx.x * blockDim.x + threadIdx.x;   // code row, 0..1023
    if (j < VQ_N_E) {
        float s = 0.0f;
        #pragma unroll 8
        for (int k = 0; k < VQ_EDIM; ++k) {
            float v = cb[j * VQ_EDIM + k];
            s += v * v;
            cb16[j * VQ_EDIM + k] = (_Float16)v;
        }
        c2[j]   = s;
        hist[j] = 0u;
    }
}

// ---------------------------------------------------------------------------
// Kernel 1: async-stage codebook to LDS, WMMA argmax of (z.c - 0.5|c|^2),
// then gather. 256 threads = 8 waves; each wave owns 32 tokens (two M-tiles
// sharing each B fragment) -> 256 tokens per block.
// ---------------------------------------------------------------------------
__global__ __launch_bounds__(256)
void vq_main(const float* __restrict__ z,
             const float* __restrict__ cb,
             const _Float16* __restrict__ cb16,
             const float* __restrict__ c2,
             unsigned* __restrict__ hist,
             float* __restrict__ blockErr,
             float* __restrict__ out) {
    extern __shared__ __align__(32) char smem[];
    _Float16* cbL  = (_Float16*)smem;
    float*    c2L  = (float*)(smem + 131072);
    int*      sIdx = (int*)(smem + 135168);
    float*    sErr = (float*)(smem + 136192);

    const int tid      = threadIdx.x;
    const int wave     = tid >> 5;
    const int lane     = tid & 31;
    const int laneHalf = lane & 15;
    const int hiK      = (lane >= 16) ? 16 : 0;  // A-fragment K sub-offset
    const int bko      = (lane >= 16) ? 16 : 0;  // B-fragment K sub-offset

    // ---- Async stage: f16 codebook (8192 x b128) + c2 (256 x b128) to LDS
    {
        const unsigned cbBase = (unsigned)(uintptr_t)cbL;
        const char*    gsrc   = (const char*)cb16;
        #pragma unroll 4
        for (int i = tid; i < 8192; i += 256) {
            unsigned           l = cbBase + (unsigned)i * 16u;
            unsigned long long g = (unsigned long long)(uintptr_t)(gsrc + (size_t)i * 16);
            asm volatile("global_load_async_to_lds_b128 %0, %1, off"
                         :: "v"(l), "v"(g) : "memory");
        }
        unsigned           l2 = (unsigned)(uintptr_t)c2L + (unsigned)tid * 16u;
        unsigned long long g2 = (unsigned long long)(uintptr_t)((const char*)c2 + (size_t)tid * 16);
        asm volatile("global_load_async_to_lds_b128 %0, %1, off"
                     :: "v"(l2), "v"(g2) : "memory");
        asm volatile("s_wait_asynccnt 0x0" ::: "memory");
    }
    __syncthreads();

    // ---- A fragments for two 16-token tiles (contiguous K order, k = 16*(lane/16) + s:
    // lane holds row M=lane%16; lane-half selects K sub-ranges)
    const long t0 = (long)blockIdx.x * 256 + wave * 32;
    v16h a0, a1, a2, a3;
    {
        const float* zp = z + (t0 + laneHalf) * VQ_EDIM;
        const float* zq = z + (t0 + 16 + laneHalf) * VQ_EDIM;
        #pragma unroll
        for (int i = 0; i < 8; ++i) {
            a0[i]     = (_Float16)zp[hiK + i];
            a0[8 + i] = (_Float16)zp[8 + hiK + i];
            a1[i]     = (_Float16)zp[32 + hiK + i];
            a1[8 + i] = (_Float16)zp[40 + hiK + i];
            a2[i]     = (_Float16)zq[hiK + i];
            a2[8 + i] = (_Float16)zq[8 + hiK + i];
            a3[i]     = (_Float16)zq[32 + hiK + i];
            a3[8 + i] = (_Float16)zq[40 + hiK + i];
        }
    }

    float best0[8], best1[8];
    int   bidx0[8], bidx1[8];
    #pragma unroll
    for (int r = 0; r < 8; ++r) {
        best0[r] = -3.4e38f; bidx0[r] = 0;
        best1[r] = -3.4e38f; bidx1[r] = 0;
    }

    // ---- Main loop: B fragments from LDS, C pre-loaded with -0.5|c|^2 so the
    // WMMA output is directly the score to maximize.
    for (int n0 = 0; n0 < VQ_N_E; n0 += 16) {
        const _Float16* bp = cbL + (n0 + laneHalf) * VQ_EDIM;
        v16h b0 = *(const v16h*)(bp + bko);        // K 0..31
        v16h b1 = *(const v16h*)(bp + 32 + bko);   // K 32..63
        const float cinit = -0.5f * c2L[n0 + laneHalf];
        const int   nIdx  = n0 + laneHalf;

        v8f acc0, acc1;
        #pragma unroll
        for (int r = 0; r < 8; ++r) { acc0[r] = cinit; acc1[r] = cinit; }

        acc0 = __builtin_amdgcn_wmma_f32_16x16x32_f16(false, a0, false, b0,
                                                      (short)0, acc0, false, false);
        acc0 = __builtin_amdgcn_wmma_f32_16x16x32_f16(false, a1, false, b1,
                                                      (short)0, acc0, false, false);
        acc1 = __builtin_amdgcn_wmma_f32_16x16x32_f16(false, a2, false, b0,
                                                      (short)0, acc1, false, false);
        acc1 = __builtin_amdgcn_wmma_f32_16x16x32_f16(false, a3, false, b1,
                                                      (short)0, acc1, false, false);

        #pragma unroll
        for (int r = 0; r < 8; ++r) {
            if (acc0[r] > best0[r]) { best0[r] = acc0[r]; bidx0[r] = nIdx; }
            if (acc1[r] > best1[r]) { best1[r] = acc1[r]; bidx1[r] = nIdx; }
        }
    }

    // ---- Cross-lane arg-max within each 16-lane half (tie -> lower index)
    #pragma unroll
    for (int r = 0; r < 8; ++r) {
        #pragma unroll
        for (int m = 1; m < 16; m <<= 1) {
            float ov = __shfl_xor(best0[r], m, 32);
            int   oi = __shfl_xor(bidx0[r], m, 32);
            if (ov > best0[r] || (ov == best0[r] && oi < bidx0[r])) {
                best0[r] = ov; bidx0[r] = oi;
            }
            float pv = __shfl_xor(best1[r], m, 32);
            int   pi = __shfl_xor(bidx1[r], m, 32);
            if (pv > best1[r] || (pv == best1[r] && pi < bidx1[r])) {
                best1[r] = pv; bidx1[r] = pi;
            }
        }
    }

    // ---- Publish: lane 0 -> rows 0..7, lane 16 -> rows 8..15 (per tile)
    const size_t Z = (size_t)VQ_NTOK * VQ_EDIM;
    if (laneHalf == 0) {
        const int rbase = (lane >= 16) ? 8 : 0;
        #pragma unroll
        for (int r = 0; r < 8; ++r) {
            int tl0 = wave * 32 + rbase + r;        // tile 0 local token
            int tl1 = tl0 + 16;                      // tile 1 local token
            sIdx[tl0] = bidx0[r];
            sIdx[tl1] = bidx1[r];
            int base = (int)(blockIdx.x) * 256;
            out[Z + 1 + base + tl0] = (float)bidx0[r];
            out[Z + 1 + base + tl1] = (float)bidx1[r];
            atomicAdd(&hist[bidx0[r]], 1u);
            atomicAdd(&hist[bidx1[r]], 1u);
        }
    }
    __syncthreads();

    // ---- Gather z_q (forward value of z_q_st) + squared error, coalesced
    const long tb = (long)blockIdx.x * 256;
    float err = 0.0f;
    #pragma unroll 4
    for (int i = 0; i < 64; ++i) {
        int e  = i * 256 + tid;      // 0..16383 (256 tokens * 64 dims)
        int tl = e >> 6;
        int k  = e & 63;
        int ci = sIdx[tl];
        float zq = cb[ci * VQ_EDIM + k];
        float zv = z[(tb + tl) * VQ_EDIM + k];
        out[(tb + tl) * VQ_EDIM + k] = zq;
        float dd = zq - zv;
        err += dd * dd;
    }
    sErr[tid] = err;
    __syncthreads();
    #pragma unroll
    for (int s = 128; s > 0; s >>= 1) {
        if (tid < s) sErr[tid] += sErr[tid + s];
        __syncthreads();
    }
    if (tid == 0) blockErr[blockIdx.x] = sErr[0];    // no float atomics
}

// ---------------------------------------------------------------------------
// Kernel 2: scalar losses (single block, deterministic reductions)
// ---------------------------------------------------------------------------
__global__ void vq_finalize(const unsigned* __restrict__ hist,
                            const float* __restrict__ blockErr,
                            float* __restrict__ out) {
    __shared__ float shU[256];
    __shared__ float shE[256];
    int tid = threadIdx.x;
    float u = 0.0f, e = 0.0f;
    #pragma unroll
    for (int i = tid; i < VQ_N_E; i += 256) {
        float p = (float)hist[i] * (1.0f / (float)VQ_NTOK);
        u += p * __logf(p + 1e-10f);
    }
    #pragma unroll
    for (int i = tid; i < 512; i += 256) e += blockErr[i];
    shU[tid] = u;
    shE[tid] = e;
    __syncthreads();
    #pragma unroll
    for (int s = 128; s > 0; s >>= 1) {
        if (tid < s) { shU[tid] += shU[tid + s]; shE[tid] += shE[tid + s]; }
        __syncthreads();
    }
    if (tid == 0) {
        const size_t Z = (size_t)VQ_NTOK * VQ_EDIM;
        float usage = -shU[0];
        float mse   = shE[0] / (float)((size_t)VQ_NTOK * VQ_EDIM);
        out[Z] = (1.0f + VQ_BETA) * mse + 0.01f * usage;   // vq_loss
        out[Z + 1 + VQ_NTOK] = __expf(usage);              // perplexity
    }
}

extern "C" void kernel_launch(void* const* d_in, const int* in_sizes, int n_in,
                              void* d_out, int out_size, void* d_ws, size_t ws_size,
                              hipStream_t stream) {
    const float* z  = (const float*)d_in[0];   // [131072, 64] f32
    const float* cb = (const float*)d_in[1];   // [1024, 64]   f32
    char* ws = (char*)d_ws;
    _Float16* cb16     = (_Float16*)(ws);
    float*    c2       = (float*)(ws + 131072);
    unsigned* hist     = (unsigned*)(ws + 135168);
    float*    blockErr = (float*)(ws + 139264);
    float*    out      = (float*)d_out;

    vq_prep<<<4, 256, 0, stream>>>(cb, cb16, c2, hist);
    vq_main<<<VQ_NTOK / 256, 256, VQ_LDS_BYTES, stream>>>(z, cb, cb16, c2, hist,
                                                          blockErr, out);
    vq_finalize<<<1, 256, 0, stream>>>(hist, blockErr, out);
}
